// GumbelQuantizer_83743272337522
// MI455X (gfx1250) — matrix operand order repaired and checked by reading, not compiled, not run on hardware
//
#include <hip/hip_runtime.h>
#include <hip/hip_bf16.h>
#include <math.h>

// ---------------- problem constants ----------------
#define KCODES 8192    // n_embeddings
#define DEMB   256     // embedding dim
#define NTOK   8192    // B*H*W = 8*32*32
#define NCHW   2097152 // 8*256*32*32
// d_out layout (float): [0..NCHW) z_q_ste, [NCHW] vq_loss, [NCHW+1] perplexity,
// [NCHW+2 .. NCHW+2+NTOK) encoding_indices (as float)
#define OUT_LOSS (NCHW)
#define OUT_PERP (NCHW + 1)
#define OUT_IDX  (NCHW + 2)

typedef __attribute__((ext_vector_type(16))) __bf16 v16bf;
typedef __attribute__((ext_vector_type(8)))  __bf16 v8bf;
typedef __attribute__((ext_vector_type(8)))  float  v8f;
typedef __attribute__((ext_vector_type(4)))  int    v4i;

#if defined(__gfx1250__) && __has_builtin(__builtin_amdgcn_global_load_async_to_lds_b128) && \
    __has_builtin(__builtin_amdgcn_s_wait_asynccnt)
#define USE_ASYNC_LDS 1
typedef __attribute__((address_space(1))) v4i as1_v4i;  // global (64-bit ptr)
typedef __attribute__((address_space(3))) v4i as3_v4i;  // LDS (32-bit ptr)
#else
#define USE_ASYNC_LDS 0
#endif

static __device__ __forceinline__ v16bf cat16(v8bf lo, v8bf hi) {
  return __builtin_shufflevector(lo, hi, 0,1,2,3,4,5,6,7,8,9,10,11,12,13,14,15);
}

// ---- kernel 1: weight fp32 -> bf16, wsum[k] = ||w_k||^2, zero counts/packed ----
__global__ __launch_bounds__(256) void k_wconv(const float* __restrict__ w,
                                               __bf16* __restrict__ Wb,
                                               float* __restrict__ wsum,
                                               int* __restrict__ counts,
                                               unsigned long long* __restrict__ packed) {
  int wave = threadIdx.x >> 5, lane = threadIdx.x & 31;
  int row  = blockIdx.x * 8 + wave;                 // 1024 blocks -> 8192 rows
  const float* src = w + (size_t)row * DEMB;
  __bf16* dst = Wb + (size_t)row * DEMB;
  float acc = 0.f;
#pragma unroll
  for (int e = 0; e < 8; ++e) {
    float x = src[lane + 32 * e];
    dst[lane + 32 * e] = (__bf16)x;
    acc += x * x;
  }
#pragma unroll
  for (int off = 16; off; off >>= 1) acc += __shfl_xor(acc, off, 32);
  if (lane == 0) { wsum[row] = acc; counts[row] = 0; packed[row] = 0ULL; }
}

// ---- kernel 2: z_e NCHW fp32 -> Zb[N][D] bf16 (LDS-tiled transpose) ----
__global__ __launch_bounds__(256) void k_zconv(const float* __restrict__ ze,
                                               __bf16* __restrict__ Zb) {
  __shared__ __bf16 tile[32][34];                   // [d_local][hw_local], padded
  int b = blockIdx.z, hw0 = blockIdx.x * 32, d0 = blockIdx.y * 32;
  int tx = threadIdx.x, ty = threadIdx.y;           // block (32, 8)
#pragma unroll
  for (int s = 0; s < 4; ++s) {
    int dl = ty + 8 * s;
    float v = __builtin_nontemporal_load(
        ze + ((size_t)(b * DEMB + d0 + dl)) * 1024 + hw0 + tx);     // coalesced, streamed
    tile[dl][tx] = (__bf16)v;
  }
  __syncthreads();
#pragma unroll
  for (int s = 0; s < 4; ++s) {
    int hwl = ty + 8 * s;
    Zb[((size_t)(b * 1024 + hw0 + hwl)) * DEMB + d0 + tx] = tile[tx][hwl]; // coalesced
  }
}

// ---- kernel 3: fused score GEMM (bf16 WMMA) + gumbel + argmax ----
// grid = 512 blocks: mb = bx>>3 (row panel of 128), ks = bx&7 (1024-col split)
// 8 waves/block; wave owns 16 rows, A panel resident in registers (8 x v16bf).
// B tiles double-buffered through LDS via gfx1250 async global->LDS copies.
// Accumulator pre-loaded with 0.5*(gumbel - ||w||^2): the 8 chained WMMAs then
// produce the final gumbel-perturbed score directly (argmax-equivalent form).
__global__ __launch_bounds__(256) void k_gemm(const __bf16* __restrict__ Zb,
                                              const __bf16* __restrict__ Wb,
                                              const float* __restrict__ wsum,
                                              const float* __restrict__ gum,
                                              unsigned long long* __restrict__ packed) {
  constexpr int LROW = 264;                          // padded LDS row (bf16 elems)
  __shared__ __bf16 bufs[2][16 * LROW];
  const int mb = blockIdx.x >> 3, ks = blockIdx.x & 7;
  const int wave = threadIdx.x >> 5, lane = threadIdx.x & 31;
  const int hi = lane >> 4, l15 = lane & 15;
  const int m0 = mb * 128 + wave * 16;
  const int c_base = ks * 1024;
  const int tid = threadIdx.x;

  // A panel: lane l15 holds row m0+l15; per k-step j (K=32j..32j+31):
  // V0-3 = K 8*hi+0..7, V4-7 = K 16+8*hi+0..7  (ISA 16-bit A 16x32 layout)
  v16bf a[8];
  const __bf16* Arow = Zb + (size_t)(m0 + l15) * DEMB + 8 * hi;
#pragma unroll
  for (int j = 0; j < 8; ++j) {
    v8bf lo = *(const v8bf*)(Arow + 32 * j);
    v8bf hp = *(const v8bf*)(Arow + 32 * j + 16);
    a[j] = cat16(lo, hp);
  }

  float bestv[8]; int besti[8];
#pragma unroll
  for (int r = 0; r < 8; ++r) { bestv[r] = -3.4e38f; besti[r] = 0; }

  // cooperative B-tile copy: 16 codebook rows x 256 bf16 -> LDS (padded rows)
  auto loadTile = [&](int t, int bsel) {
    int c0 = c_base + t * 16;
#pragma unroll
    for (int q = 0; q < 2; ++q) {
      int cch = tid + q * 256;                       // 512 chunks of 8 bf16
      int rw = cch >> 5, sg = cch & 31;
      const __bf16* g = Wb + (size_t)(c0 + rw) * DEMB + sg * 8;
      __bf16* l = &bufs[bsel][rw * LROW + sg * 8];
#if USE_ASYNC_LDS
      __builtin_amdgcn_global_load_async_to_lds_b128(
          (as1_v4i*)(uintptr_t)(const void*)g,
          (as3_v4i*)(unsigned int)(uintptr_t)(void*)l, 0, 0);
#else
      *(uint4*)l = *(const uint4*)g;
#endif
    }
  };
  auto waitTile = [&]() {
#if USE_ASYNC_LDS
    __builtin_amdgcn_s_wait_asynccnt(0);
#endif
  };

  loadTile(0, 0);
  waitTile();
  __syncthreads();

  for (int t = 0; t < 64; ++t) {
    const int cur = t & 1;
    if (t + 1 < 64) loadTile(t + 1, cur ^ 1);        // overlap copy with compute
    const int n0 = c_base + t * 16;
    const int col = n0 + l15;
    const float ws = wsum[col];
    v8f c;
#pragma unroll
    for (int r = 0; r < 8; ++r) {                    // C[r]: row m0+r+8*hi, col n0+l15
      float g = __builtin_nontemporal_load(
          gum + (size_t)(m0 + r + 8 * hi) * KCODES + col); // 256MB one-shot stream
      c[r] = 0.5f * (g - ws);
    }
    // B operand 32x16 (KxN): lane = col n, holds W[n0+l15][32j + 8*hi + 0..7, 32j + 16 + 8*hi + 0..7]
    const __bf16* brow = &bufs[cur][l15 * LROW + 8 * hi];
#pragma unroll
    for (int j = 0; j < 8; ++j) {
      v8bf blo = *(const v8bf*)(brow + 32 * j);
      v8bf bhi = *(const v8bf*)(brow + 32 * j + 16);
      v16bf bb = cat16(blo, bhi);
      c = __builtin_amdgcn_wmma_f32_16x16x32_bf16(false, a[j], false, bb,
                                                  (short)0, c, false, false);
    }
#pragma unroll
    for (int r = 0; r < 8; ++r)
      if (c[r] > bestv[r]) { bestv[r] = c[r]; besti[r] = col; } // strict >: ascending cols => min-index ties
    waitTile();
    __syncthreads();
  }

  // reduce across each 16-lane half (rows replicated across the half)
#pragma unroll
  for (int r = 0; r < 8; ++r) {
    float v = bestv[r]; int i = besti[r];
#pragma unroll
    for (int off = 8; off; off >>= 1) {
      float vo = __shfl_xor(v, off, 32);
      int   io = __shfl_xor(i, off, 32);
      if (vo > v || (vo == v && io < i)) { v = vo; i = io; }
    }
    if (l15 == 0) {
      int row = m0 + r + 8 * hi;
      unsigned int ub = __float_as_uint(v);
      ub = (ub & 0x80000000u) ? ~ub : (ub | 0x80000000u);      // order-preserving map
      unsigned long long pk =
          ((unsigned long long)ub << 32) | (unsigned int)(~i); // ~i: min idx wins ties
      atomicMax(&packed[row], pk);
    }
  }
}

// ---- kernel 4: unpack argmax, build usage counts ----
__global__ __launch_bounds__(256) void k_extract(const unsigned long long* __restrict__ packed,
                                                 int* __restrict__ final_idx,
                                                 int* __restrict__ counts) {
  int n = blockIdx.x * 256 + threadIdx.x;
  unsigned long long p = packed[n];
  int idx = (int)(~(unsigned int)(p & 0xffffffffu));
  final_idx[n] = idx;
  atomicAdd(&counts[idx], 1);
}

// ---- kernel 5: z_q gather (fp32), NCHW store, per-block loss partials ----
__global__ __launch_bounds__(256) void k_quant(const float* __restrict__ ze,
                                               const float* __restrict__ w,
                                               const int* __restrict__ final_idx,
                                               float* __restrict__ out,
                                               float* __restrict__ partial) {
  __shared__ float red[256];
  int i = blockIdx.x * 256 + threadIdx.x;           // NCHW flat index
  int hw = i & 1023, d = (i >> 10) & 255, b = i >> 18;
  int n = b * 1024 + hw;
  int code = final_idx[n];
  float q = w[(size_t)code * DEMB + d];             // codebook: keep cached (L2-hot)
  __builtin_nontemporal_store(q, out + i);          // z_q_ste == z_q (forward), write-once
  float diff = q - __builtin_nontemporal_load(ze + i);
  red[threadIdx.x] = diff * diff;
  __syncthreads();
  for (int s = 128; s; s >>= 1) {
    if (threadIdx.x < s) red[threadIdx.x] += red[threadIdx.x + s];
    __syncthreads();
  }
  if (threadIdx.x == 0) partial[blockIdx.x] = red[0];
}

// ---- kernel 6: scalars + index copy (single block, fixed-order => deterministic) ----
__global__ __launch_bounds__(256) void k_final(const float* __restrict__ partial,
                                               const int* __restrict__ counts,
                                               const int* __restrict__ final_idx,
                                               float* __restrict__ out) {
  __shared__ float red[256];
  int tid = threadIdx.x;
  float ls = 0.f;
  for (int i = tid; i < 8192; i += 256) ls += partial[i];
  red[tid] = ls; __syncthreads();
  for (int s = 128; s; s >>= 1) { if (tid < s) red[tid] += red[tid + s]; __syncthreads(); }
  float lsum = red[0];
  __syncthreads();
  float hs = 0.f;
  for (int i = tid; i < KCODES; i += 256) {
    float p = (float)counts[i] * (1.0f / (float)NTOK);
    hs += p * logf(p + 1e-10f);
  }
  red[tid] = hs; __syncthreads();
  for (int s = 128; s; s >>= 1) { if (tid < s) red[tid] += red[tid + s]; __syncthreads(); }
  if (tid == 0) {
    out[OUT_LOSS] = 1.25f * lsum / (float)NCHW;     // (1+BETA)*MSE
    out[OUT_PERP] = expf(-red[0]);
  }
  for (int i = tid; i < NTOK; i += 256) out[OUT_IDX + i] = (float)final_idx[i];
}

extern "C" void kernel_launch(void* const* d_in, const int* in_sizes, int n_in,
                              void* d_out, int out_size, void* d_ws, size_t ws_size,
                              hipStream_t stream) {
  const float* z_e    = (const float*)d_in[0];   // [8,256,32,32]
  const float* weight = (const float*)d_in[1];   // [8192,256]
  const float* gumbel = (const float*)d_in[2];   // [8192,8192]
  float* out = (float*)d_out;

  // workspace layout (256B-aligned regions)
  char* ws = (char*)d_ws;
  __bf16* Zb               = (__bf16*)(ws + 0);                       // 4 MiB
  __bf16* Wb               = (__bf16*)(ws + 4194304);                 // 4 MiB
  float*  wsum             = (float*)(ws + 8388608);                  // 32 KiB
  unsigned long long* pckd = (unsigned long long*)(ws + 8421376);     // 64 KiB
  int*    final_idx        = (int*)(ws + 8486912);                    // 32 KiB
  int*    counts           = (int*)(ws + 8519680);                    // 32 KiB
  float*  partial          = (float*)(ws + 8552448);                  // 32 KiB

  k_wconv<<<1024, 256, 0, stream>>>(weight, Wb, wsum, counts, pckd);
  k_zconv<<<dim3(32, 8, 8), dim3(32, 8), 0, stream>>>(z_e, Zb);
  k_gemm<<<512, 256, 0, stream>>>(Zb, Wb, wsum, gumbel, pckd);
  k_extract<<<32, 256, 0, stream>>>(pckd, final_idx, counts);
  k_quant<<<8192, 256, 0, stream>>>(z_e, weight, final_idx, out, partial);
  k_final<<<1, 256, 0, stream>>>(partial, counts, final_idx, out);
}
